// StyleHyperLinear_60000693125372
// MI455X (gfx1250) — matrix operand order repaired and checked by reading, not compiled, not run on hardware
//
#include <hip/hip_runtime.h>
#include <hip/hip_bf16.h>

// ---------------------------------------------------------------------------
// StyleHyperLinear on MI455X (gfx1250, wave32, WMMA + Tensor Data Mover).
//
// out = x @ (w_base + mat_b @ mat_a)^T + b_base      (LoRA folded into GEMM)
//
// Pipeline:
//   1) hyper_kernel : hypernet -> raw LoRA params (fp32, ws)
//   2) weff_kernel  : W_eff = w_base + mat_b@mat_a, split to bf16 hi/lo (ws)
//   3) xsplit_kernel: x fp32 -> bf16 hi/lo arrays (ws)  [one-shot, so the GEMM
//                     inner loop has ZERO conversion VALU]
//   4) gemm_kernel  : bf16x3 WMMA GEMM, tiles staged by TDM tensor_load_to_lds
//                     with double buffering + s_wait_tensorcnt.
// ---------------------------------------------------------------------------

#define B_DIM   8
#define T_DIM   4096
#define D_IN    512
#define D_OUT   512
#define STYLE   256
#define RANK    8
#define NUM_LORA ((D_IN + D_OUT) * RANK)   // 8192

typedef __attribute__((ext_vector_type(16))) __bf16 v16bf;
typedef __attribute__((ext_vector_type(8)))  float  v8f;

typedef __attribute__((ext_vector_type(4))) unsigned int tdm_v4u;
typedef __attribute__((ext_vector_type(8))) int          tdm_v8i;
typedef __attribute__((ext_vector_type(4))) int          tdm_v4i;

#if __has_builtin(__builtin_amdgcn_tensor_load_to_lds)
#define USE_TDM 1
#else
#define USE_TDM 0
#endif

// ---- bf16 helpers (round-to-nearest-even via bit ops) ----------------------
__device__ __forceinline__ unsigned short f2bf(float f) {
    unsigned u = __float_as_uint(f);
    unsigned r = u + 0x7FFFu + ((u >> 16) & 1u);
    return (unsigned short)(r >> 16);
}
__device__ __forceinline__ float bf2f(unsigned short h) {
    return __uint_as_float(((unsigned)h) << 16);
}
__device__ __forceinline__ void split_bf(float f, unsigned short& hi, unsigned short& lo) {
    hi = f2bf(f);
    lo = f2bf(f - bf2f(hi));
}

// Generic pointer to LDS byte offset: ISA 10.2 — flat addresses in the LDS
// aperture carry the allocation-relative LDS offset in addr[31:0].
__device__ __forceinline__ unsigned lds_off(const void* p) {
    return (unsigned)(size_t)p;
}

// ---------------------------------------------------------------------------
// Kernel 1: hypernet.  h = silu(style @ w_h1^T + b_h1); raw = h @ w_h2^T + b_h2
// ---------------------------------------------------------------------------
__global__ __launch_bounds__(256)
void hyper_kernel(const float* __restrict__ style,
                  const float* __restrict__ w_h1, const float* __restrict__ b_h1,
                  const float* __restrict__ w_h2, const float* __restrict__ b_h2,
                  float* __restrict__ raw) {
    __shared__ float s_style[STYLE];
    __shared__ float s_h[STYLE];
    const int b = blockIdx.x;
    const int t = threadIdx.x;             // 256 threads

    s_style[t] = style[b * STYLE + t];
    __syncthreads();

    {
        float acc = b_h1[t];
        const float* w1 = w_h1 + (size_t)t * STYLE;
        #pragma unroll 8
        for (int d = 0; d < STYLE; ++d) acc += s_style[d] * w1[d];
        s_h[t] = acc / (1.0f + __expf(-acc));
    }
    __syncthreads();

    for (int j = t; j < NUM_LORA; j += 256) {
        float acc = b_h2[j];
        const float* w2 = w_h2 + (size_t)j * STYLE;
        #pragma unroll 8
        for (int d = 0; d < STYLE; ++d) acc += s_h[d] * w2[d];
        raw[(size_t)b * NUM_LORA + j] = acc;
    }
}

// ---------------------------------------------------------------------------
// Kernel 2: W_eff[b][o][d] = w_base[o][d] + sum_r mat_b[b][o][r]*mat_a[b][r][d]
// split to bf16 hi/lo.
// ---------------------------------------------------------------------------
__global__ __launch_bounds__(256)
void weff_kernel(const float* __restrict__ w_base,
                 const float* __restrict__ raw,
                 unsigned short* __restrict__ w_hi,
                 unsigned short* __restrict__ w_lo) {
    const int idx = blockIdx.x * 256 + threadIdx.x;     // < B*D_OUT*D_IN = 2^21
    const int d = idx & (D_IN - 1);
    const int o = (idx >> 9) & (D_OUT - 1);
    const int b = idx >> 18;

    const float* mat_a = raw + (size_t)b * NUM_LORA;                  // [RANK][D_IN]
    const float* mat_b = raw + (size_t)b * NUM_LORA + D_IN * RANK;    // [D_OUT][RANK]

    float w = w_base[(size_t)o * D_IN + d];
    #pragma unroll
    for (int r = 0; r < RANK; ++r)
        w += mat_b[o * RANK + r] * mat_a[r * D_IN + d];

    unsigned short hi, lo;
    split_bf(w, hi, lo);
    const size_t off = ((size_t)b * D_OUT + o) * D_IN + d;
    w_hi[off] = hi;
    w_lo[off] = lo;
}

// ---------------------------------------------------------------------------
// Kernel 3: x fp32 -> bf16 hi/lo (4 elements / thread, vectorized)
// ---------------------------------------------------------------------------
__global__ __launch_bounds__(256)
void xsplit_kernel(const float* __restrict__ x,
                   unsigned short* __restrict__ xh,
                   unsigned short* __restrict__ xl) {
    const size_t i0 = ((size_t)blockIdx.x * 256 + threadIdx.x) * 4;
    const float4 v = *(const float4*)(x + i0);
    unsigned short h0, h1, h2, h3, l0, l1, l2, l3;
    split_bf(v.x, h0, l0); split_bf(v.y, h1, l1);
    split_bf(v.z, h2, l2); split_bf(v.w, h3, l3);
    uint2 ph, pl;
    ph.x = (unsigned)h0 | ((unsigned)h1 << 16);
    ph.y = (unsigned)h2 | ((unsigned)h3 << 16);
    pl.x = (unsigned)l0 | ((unsigned)l1 << 16);
    pl.y = (unsigned)l2 | ((unsigned)l3 << 16);
    *(uint2*)(xh + i0) = ph;
    *(uint2*)(xl + i0) = pl;
}

// ---------------------------------------------------------------------------
// Kernel 4: main GEMM, bf16x3 WMMA, TDM-staged double-buffered LDS.
// WG = 256 threads = 8 waves.  Block tile 128(M) x 128(N), K chunks of 32.
// Wave w owns rows [w*16, w*16+16) and all 128 N columns -> 8 C tiles,
// 24 v_wmma_f32_16x16x32_bf16 per K chunk (hi*hi + hi*lo + lo*hi).
// ---------------------------------------------------------------------------
#define BM 128
#define BN 128
#define BK 32
#define NIT (D_IN / BK)     // 16 K chunks
#define NT  (BN / 16)       // 8 C tiles per wave

union Frag {
    v16bf v;
    uint4 q[2];
};

#if USE_TDM
// Issue a 2D TDM tile load: tile1 x tile0 bf16 elements, row stride stride0,
// into LDS at lds_byte_off (row-major contiguous).  D# packed per ISA ch.8.
__device__ __forceinline__ void tdm_load_tile(const unsigned short* g,
                                              unsigned lds_byte_off,
                                              unsigned tdim0, unsigned tdim1,
                                              unsigned tile0, unsigned tile1,
                                              unsigned stride0) {
    const unsigned long long ga = (unsigned long long)(size_t)g;
    tdm_v4u g0;
    g0.x = 1u;                                     // count=1, user descriptor
    g0.y = lds_byte_off;                           // lds_addr
    g0.z = (unsigned)ga;                           // global_addr[31:0]
    g0.w = (unsigned)(ga >> 32) | (2u << 30);      // global_addr[56:32] | type=2
    tdm_v8i g1;
    g1[0] = 0x00010000;                            // data_size=1 (2 bytes)
    g1[1] = (int)((tdim0 & 0xFFFFu) << 16);        // tensor_dim0[15:0]
    g1[2] = (int)((tdim0 >> 16) | ((tdim1 & 0xFFFFu) << 16));
    g1[3] = (int)((tdim1 >> 16) | (tile0 << 16));  // tensor_dim1 hi | tile_dim0
    g1[4] = (int)tile1;                            // tile_dim1 (tile_dim2 = 0)
    g1[5] = (int)stride0;                          // tensor_dim0_stride[31:0]
    g1[6] = 0;
    g1[7] = 0;
    tdm_v4i g2 = {0, 0, 0, 0};
    tdm_v4i g3 = {0, 0, 0, 0};
#if defined(__clang_major__) && (__clang_major__ >= 23)
    tdm_v8i g4 = {0, 0, 0, 0, 0, 0, 0, 0};
    __builtin_amdgcn_tensor_load_to_lds(g0, g1, g2, g3, g4, 0);
#else
    __builtin_amdgcn_tensor_load_to_lds(g0, g1, g2, g3, 0);
#endif
}
#endif

__global__ __launch_bounds__(256)
void gemm_kernel(const unsigned short* __restrict__ xh,
                 const unsigned short* __restrict__ xl,
                 const unsigned short* __restrict__ wh,
                 const unsigned short* __restrict__ wl,
                 const float* __restrict__ b_base,
                 float* __restrict__ out) {
    __shared__ __align__(16) unsigned short s_xh[2][BM * BK];   // 2 x 8 KB
    __shared__ __align__(16) unsigned short s_xl[2][BM * BK];   // 2 x 8 KB
    __shared__ __align__(16) unsigned short s_wh[2][BN * BK];   // 2 x 8 KB
    __shared__ __align__(16) unsigned short s_wl[2][BN * BK];   // 2 x 8 KB

    const int row0 = blockIdx.x * BM;       // flattened (b*T + t) row base
    const int n0   = blockIdx.y * BN;
    const int b    = row0 / T_DIM;          // BM | T, so block is in one batch

    const int tid  = threadIdx.x;
    const int wave = tid >> 5;              // 0..7
    const int lane = tid & 31;
    const int lsel = lane >> 4;
    const int ml   = lane & 15;

    const unsigned short* whB = wh + ((size_t)(b * D_OUT + n0)) * D_IN;
    const unsigned short* wlB = wl + ((size_t)(b * D_OUT + n0)) * D_IN;
    const unsigned short* xhB = xh + (size_t)row0 * D_IN;
    const unsigned short* xlB = xl + (size_t)row0 * D_IN;

    v8f c[NT];
    #pragma unroll
    for (int nt = 0; nt < NT; ++nt) c[nt] = (v8f){};

#if USE_TDM
    // Prologue: wave 0 kicks off the first 4 tile DMAs.
    if (wave == 0) {
        const unsigned kc = 0;
        tdm_load_tile(xhB + kc, lds_off(&s_xh[0][0]), D_IN, B_DIM * T_DIM,  BK, BM, D_IN);
        tdm_load_tile(xlB + kc, lds_off(&s_xl[0][0]), D_IN, B_DIM * T_DIM,  BK, BM, D_IN);
        tdm_load_tile(whB + kc, lds_off(&s_wh[0][0]), D_IN, B_DIM * D_OUT, BK, BN, D_IN);
        tdm_load_tile(wlB + kc, lds_off(&s_wl[0][0]), D_IN, B_DIM * D_OUT, BK, BN, D_IN);
    }
#endif

    for (int it = 0; it < NIT; ++it) {
        const int nb = it & 1;

#if USE_TDM
        if (wave == 0) {
            if (it + 1 < NIT) {
                // Prefetch next chunk into the other buffer, then wait for the
                // 4 older DMAs (current buffer) to land: TENSORcnt <= 4.
                const unsigned kc = (unsigned)(it + 1) * BK;
                const int pb = nb ^ 1;
                tdm_load_tile(xhB + kc, lds_off(&s_xh[pb][0]), D_IN, B_DIM * T_DIM,  BK, BM, D_IN);
                tdm_load_tile(xlB + kc, lds_off(&s_xl[pb][0]), D_IN, B_DIM * T_DIM,  BK, BM, D_IN);
                tdm_load_tile(whB + kc, lds_off(&s_wh[pb][0]), D_IN, B_DIM * D_OUT, BK, BN, D_IN);
                tdm_load_tile(wlB + kc, lds_off(&s_wl[pb][0]), D_IN, B_DIM * D_OUT, BK, BN, D_IN);
                __builtin_amdgcn_s_wait_tensorcnt((short)4);
            } else {
                __builtin_amdgcn_s_wait_tensorcnt((short)0);
            }
        }
        __syncthreads();     // publish TDM-written tiles to all waves
#else
        // Fallback staging: cooperative vector copies (no conversion needed,
        // operands are pre-split bf16 in global memory).
        __syncthreads();
        {
            const unsigned kc = (unsigned)it * BK;
            #pragma unroll
            for (int i = 0; i < 2; ++i) {
                const int idx = tid + i * 256;     // uint4 slot, 512 per tile
                const int r = idx >> 2;            // tile row
                const int q = idx & 3;             // uint4 within row
                ((uint4*)s_xh[nb])[idx] = ((const uint4*)(xhB + (size_t)r * D_IN + kc))[q];
                ((uint4*)s_xl[nb])[idx] = ((const uint4*)(xlB + (size_t)r * D_IN + kc))[q];
                ((uint4*)s_wh[nb])[idx] = ((const uint4*)(whB + (size_t)r * D_IN + kc))[q];
                ((uint4*)s_wl[nb])[idx] = ((const uint4*)(wlB + (size_t)r * D_IN + kc))[q];
            }
        }
        __syncthreads();
#endif

        // ---- A fragments (16x32 bf16 ISA layout): lanes 0-15 -> K 0-7,16-23;
        //      lanes 16-31 -> K 8-15,24-31 ----
        const uint4* axh = (const uint4*)(&s_xh[nb][(wave * 16 + ml) * BK]);
        const uint4* axl = (const uint4*)(&s_xl[nb][(wave * 16 + ml) * BK]);
        Frag ah, al;
        ah.q[0] = axh[lsel];  ah.q[1] = axh[lsel + 2];
        al.q[0] = axl[lsel];  al.q[1] = axl[lsel + 2];

        #pragma unroll
        for (int nt = 0; nt < NT; ++nt) {
            // B fragment (32x16): lanes 0-15 -> K 0-7,16-23, lanes 16-31 -> K 8-15,24-31
            const uint4* bwh = (const uint4*)(&s_wh[nb][(nt * 16 + ml) * BK]);
            const uint4* bwl = (const uint4*)(&s_wl[nb][(nt * 16 + ml) * BK]);
            Frag bh, bl;
            bh.q[0] = bwh[lsel];  bh.q[1] = bwh[lsel + 2];
            bl.q[0] = bwl[lsel];  bl.q[1] = bwl[lsel + 2];

            c[nt] = __builtin_amdgcn_wmma_f32_16x16x32_bf16(
                        false, ah.v, false, bh.v, (short)0, c[nt], false, false);
            c[nt] = __builtin_amdgcn_wmma_f32_16x16x32_bf16(
                        false, ah.v, false, bl.v, (short)0, c[nt], false, false);
            c[nt] = __builtin_amdgcn_wmma_f32_16x16x32_bf16(
                        false, al.v, false, bh.v, (short)0, c[nt], false, false);
        }
        __syncthreads();     // all waves done with buffer nb before it refills
    }

    // ---- epilogue: C layout vgpr j -> M = j + 8*lsel, N = ml ----
    #pragma unroll
    for (int nt = 0; nt < NT; ++nt) {
        const int n = n0 + nt * 16 + ml;
        const float bias = b_base[n];
        #pragma unroll
        for (int j = 0; j < 8; ++j) {
            const int m = wave * 16 + j + 8 * lsel;
            out[(size_t)(row0 + m) * D_OUT + n] = c[nt][j] + bias;
        }
    }
}

// ---------------------------------------------------------------------------
extern "C" void kernel_launch(void* const* d_in, const int* in_sizes, int n_in,
                              void* d_out, int out_size, void* d_ws, size_t ws_size,
                              hipStream_t stream) {
    const float* x         = (const float*)d_in[0];
    const float* style_emb = (const float*)d_in[1];
    const float* w_base    = (const float*)d_in[2];
    const float* b_base    = (const float*)d_in[3];
    const float* w_h1      = (const float*)d_in[4];
    const float* b_h1      = (const float*)d_in[5];
    const float* w_h2      = (const float*)d_in[6];
    const float* b_h2      = (const float*)d_in[7];
    float* out = (float*)d_out;

    // workspace layout (bytes):
    //   raw  : B*NUM_LORA fp32            =  256 KB
    //   w_hi : B*D_OUT*D_IN bf16          =    4 MB
    //   w_lo : B*D_OUT*D_IN bf16          =    4 MB
    //   x_hi : B*T*D_IN bf16              =   32 MB
    //   x_lo : B*T*D_IN bf16              =   32 MB
    char* ws = (char*)d_ws;
    float* raw = (float*)ws;
    size_t off = (size_t)B_DIM * NUM_LORA * sizeof(float);
    unsigned short* w_hi = (unsigned short*)(ws + off);
    off += (size_t)B_DIM * D_OUT * D_IN * sizeof(unsigned short);
    unsigned short* w_lo = (unsigned short*)(ws + off);
    off += (size_t)B_DIM * D_OUT * D_IN * sizeof(unsigned short);
    unsigned short* x_hi = (unsigned short*)(ws + off);
    off += (size_t)B_DIM * T_DIM * D_IN * sizeof(unsigned short);
    unsigned short* x_lo = (unsigned short*)(ws + off);

    hyper_kernel<<<B_DIM, 256, 0, stream>>>(style_emb, w_h1, b_h1, w_h2, b_h2, raw);

    weff_kernel<<<(B_DIM * D_OUT * D_IN) / 256, 256, 0, stream>>>(w_base, raw, w_hi, w_lo);

    xsplit_kernel<<<(B_DIM * T_DIM * D_IN) / (256 * 4), 256, 0, stream>>>(x, x_hi, x_lo);

    dim3 grid((B_DIM * T_DIM) / BM, D_OUT / BN);
    gemm_kernel<<<grid, 256, 0, stream>>>(x_hi, x_lo, w_hi, w_lo, b_base, out);
}
